// GPT_12146167513120
// MI455X (gfx1250) — matrix operand order repaired and checked by reading, not compiled, not run on hardware
//
#include <hip/hip_runtime.h>
#include <cstdint>

// ---------------------------------------------------------------------------
// MI455X (gfx1250) GPT forward: bf16 WMMA GEMMs + flash attention.
// Weights are pre-transposed/converted to bf16 once per layer; GEMM inner
// loop is a single-barrier, fully async double-buffered WMMA pipeline.
// ---------------------------------------------------------------------------

typedef __attribute__((ext_vector_type(16))) __bf16 v16bf;
typedef __attribute__((ext_vector_type(8)))  float  v8f;

union FragB16 { v16bf v; uint4 q[2]; };
static_assert(sizeof(v16bf) == 32, "v16bf must be 32 bytes");

__device__ __forceinline__ v8f wmma_bf16(v16bf a, v16bf b, v8f c) {
  return __builtin_amdgcn_wmma_f32_16x16x32_bf16(
      /*neg_a=*/false, a, /*neg_b=*/false, b,
      /*c_mod=*/(short)0, c, /*reuse_a=*/false, /*reuse_b=*/false);
}

__device__ __forceinline__ float gelu_tanh(float x) {
  float x3 = x * x * x;
  return 0.5f * x * (1.f + tanhf(0.7978845608028654f * (x + 0.044715f * x3)));
}

// Flat shared-aperture pointer -> LDS byte offset (low 32 bits of the address).
__device__ __forceinline__ uint32_t lds_off(const void* p) {
  return (uint32_t)(uintptr_t)p;
}

// Async copy 64B / 32B global -> LDS (GLOBAL_LOAD_ASYNC_TO_LDS_B128, GV mode;
// instruction offset applies to both LDS and global address per ISA 10.4.4).
__device__ __forceinline__ void async_copy64(const void* g, void* l) {
  uint32_t lo = lds_off(l);
  uint64_t ga = (uint64_t)(uintptr_t)g;
  asm volatile("global_load_async_to_lds_b128 %0, %1, off\n\t"
               "global_load_async_to_lds_b128 %0, %1, off offset:16\n\t"
               "global_load_async_to_lds_b128 %0, %1, off offset:32\n\t"
               "global_load_async_to_lds_b128 %0, %1, off offset:48"
               :: "v"(lo), "v"(ga) : "memory");
}
__device__ __forceinline__ void async_copy32(const void* g, void* l) {
  uint32_t lo = lds_off(l);
  uint64_t ga = (uint64_t)(uintptr_t)g;
  asm volatile("global_load_async_to_lds_b128 %0, %1, off\n\t"
               "global_load_async_to_lds_b128 %0, %1, off offset:16"
               :: "v"(lo), "v"(ga) : "memory");
}
__device__ __forceinline__ void wait_async0() {
  asm volatile("s_wait_asynccnt 0x0" ::: "memory");
}

// ---------------------------------------------------------------------------
// Weight pre-pass: W fp32 [K,N] -> Wt bf16 [N,K] (transpose + convert).
// One 64x64 tile per 256-thread block, staged through LDS.
// ---------------------------------------------------------------------------
__global__ __launch_bounds__(256)
void wtrans_kernel(const float* __restrict__ W, __bf16* __restrict__ Wt,
                   int K, int N)
{
  __shared__ __bf16 t[64 * 72];
  const int k0 = blockIdx.y * 64, n0 = blockIdx.x * 64;
  const int tid = threadIdx.x;
  const int kk = tid >> 4;          // 0..15
  const int nn = (tid & 15) * 4;    // 0..60
  #pragma unroll
  for (int i = 0; i < 4; ++i) {
    float4 f = *(const float4*)(W + (size_t)(k0 + kk + 16*i) * N + n0 + nn);
    t[(nn + 0) * 72 + kk + 16*i] = (__bf16)f.x;
    t[(nn + 1) * 72 + kk + 16*i] = (__bf16)f.y;
    t[(nn + 2) * 72 + kk + 16*i] = (__bf16)f.z;
    t[(nn + 3) * 72 + kk + 16*i] = (__bf16)f.w;
  }
  __syncthreads();
  const int n   = tid >> 2;         // 0..63
  const int seg = (tid & 3) * 16;   // halves
  uint4 a = *(const uint4*)&t[n * 72 + seg];
  uint4 b = *(const uint4*)&t[n * 72 + seg + 8];
  *(uint4*)(Wt + (size_t)(n0 + n) * K + k0 + seg)     = a;
  *(uint4*)(Wt + (size_t)(n0 + n) * K + k0 + seg + 8) = b;
}

// ---------------------------------------------------------------------------
// GEMM: out = act(A(bf16)[M,K] @ Wt(bf16)[N,K]^T + bias) (+ residual)
// 256 thr (8 waves), tile 128x128, BK=64 (16 WMMA / iteration).
// Both A and W tiles async double-buffered in dynamic LDS; one barrier/iter.
// ---------------------------------------------------------------------------
#define LDT 72  // LDS row pitch in halves (64 + 8 pad; 144B, 16B-aligned)
#define TILE_HALVES (128 * LDT)

template<bool GELU, bool RES, bool OUT32, bool OUT16>
__global__ __launch_bounds__(256)
void gemm_kernel(const __bf16* __restrict__ A, const __bf16* __restrict__ Wt,
                 const float* __restrict__ bias, const float* __restrict__ resid,
                 float* __restrict__ out32, __bf16* __restrict__ out16,
                 int M, int N, int K)
{
  extern __shared__ __bf16 smem[];   // 4 x TILE_HALVES = 73728 bytes
  // buffers: A tiles at smem + {0,1}*TILE_HALVES, W tiles at + {2,3}*TILE_HALVES
  // (computed per-use; pointer arrays over dynamic LDS break the linker)

  const int tid  = threadIdx.x;
  const int lane = tid & 31;
  const int wv   = tid >> 5;
  const int hl   = lane >> 4;
  const int l16  = lane & 15;
  const int m0   = blockIdx.y * 128;
  const int n0   = blockIdx.x * 128;
  const int wrow = (wv >> 2) * 64;
  const int wcol = (wv & 3) * 32;

  const v8f vz = {0.f,0.f,0.f,0.f,0.f,0.f,0.f,0.f};
  v8f acc[4][2];
  #pragma unroll
  for (int i = 0; i < 4; ++i)
    #pragma unroll
    for (int j = 0; j < 2; ++j) acc[i][j] = vz;

  float bval[2];
  #pragma unroll
  for (int j = 0; j < 2; ++j) bval[j] = bias[n0 + wcol + j*16 + l16];

  const int arow = tid >> 1;        // 0..127 (row of A tile / row of Wt tile)
  const int aseg = (tid & 1) * 32;  // halves (64B per thread)

  // ---- prologue: tile 0 ----
  async_copy64(A  + (size_t)(m0 + arow) * K + aseg,
               smem + arow*LDT + aseg);
  async_copy64(Wt + (size_t)(n0 + arow) * K + aseg,
               smem + 2*TILE_HALVES + arow*LDT + aseg);
  wait_async0();
  __syncthreads();

  for (int kt = 0; kt < K; kt += 64) {
    const int  cur = (kt >> 6) & 1;
    const bool nxt = (kt + 64) < K;
    if (nxt) {  // async prefetch next tiles into the other buffers
      async_copy64(A  + (size_t)(m0 + arow) * K + (kt + 64) + aseg,
                   smem + (cur ^ 1)*TILE_HALVES + arow*LDT + aseg);
      async_copy64(Wt + (size_t)(n0 + arow) * K + (kt + 64) + aseg,
                   smem + (2 + (cur ^ 1))*TILE_HALVES + arow*LDT + aseg);
    }
    const __bf16* Ac = smem + cur*TILE_HALVES;
    const __bf16* Wc = smem + (2 + cur)*TILE_HALVES;
    // ---- 16 WMMAs from current buffers ----
    #pragma unroll
    for (int kk = 0; kk < 2; ++kk) {
      FragB16 af[4], bf[2];
      #pragma unroll
      for (int mi = 0; mi < 4; ++mi) {  // A layout: K0-7/16-23 | K8-15/24-31
        const __bf16* p = Ac + (wrow + mi*16 + l16) * LDT + kk*32;
        af[mi].q[0] = *(const uint4*)(p + hl*8);
        af[mi].q[1] = *(const uint4*)(p + 16 + hl*8);
      }
      #pragma unroll
      for (int nj = 0; nj < 2; ++nj) {  // B layout: K0-7/16-23 | K8-15/24-31
        const __bf16* p = Wc + (wcol + nj*16 + l16) * LDT + kk*32;
        bf[nj].q[0] = *(const uint4*)(p + hl*8);
        bf[nj].q[1] = *(const uint4*)(p + 16 + hl*8);
      }
      #pragma unroll
      for (int mi = 0; mi < 4; ++mi)
        #pragma unroll
        for (int nj = 0; nj < 2; ++nj)
          acc[mi][nj] = wmma_bf16(af[mi].v, bf[nj].v, acc[mi][nj]);
    }
    if (nxt) wait_async0();     // next tiles landed
    __syncthreads();            // flip buffers
  }

  // epilogue: C/D layout is (M = r + 8*hl, N = l16) per VGPR r
  #pragma unroll
  for (int mi = 0; mi < 4; ++mi)
    #pragma unroll
    for (int nj = 0; nj < 2; ++nj)
      #pragma unroll
      for (int r = 0; r < 8; ++r) {
        int grow = m0 + wrow + mi*16 + r + hl*8;
        int gcol = n0 + wcol + nj*16 + l16;
        size_t gi = (size_t)grow * N + gcol;
        float v = acc[mi][nj][r] + bval[nj];
        if (GELU)  v = gelu_tanh(v);
        if (RES)   v += resid[gi];
        if (OUT32) out32[gi] = v;
        if (OUT16) out16[gi] = (__bf16)v;
      }
}

// ---------------------------------------------------------------------------
// Flash attention: block = (b, h, 128 q rows), 8 waves x 16 q rows each.
// qkv: bf16 [B*T, 3C]; q at col h*64, k at 1024 + h*64, v at 2048 + h*64.
// ---------------------------------------------------------------------------
__global__ __launch_bounds__(256)
void attn_kernel(const __bf16* __restrict__ qkv, const float* __restrict__ amask,
                 __bf16* __restrict__ ybuf)
{
  __shared__ __bf16 Ks[64 * 72];        // [kpos][d]
  __shared__ __bf16 Vs[64 * 72];        // transposed [d][kpos]
  __shared__ __bf16 Ss[8][16 * 72];     // per-wave P strip [qrow][kpos]

  const int tid = threadIdx.x, lane = tid & 31, wv = tid >> 5;
  const int hl = lane >> 4, l16 = lane & 15;
  const int bh = blockIdx.y;
  const int b = bh >> 4, h = bh & 15;
  const int q_base = blockIdx.x * 128;
  const int q0 = q_base + wv * 16;
  const size_t rowbase = (size_t)b * 2048 * 3072;

  // Q fragments straight from global (A layout over D=64 -> 2 k-steps)
  FragB16 qa[2];
  {
    const __bf16* qp = qkv + rowbase + (size_t)(q0 + l16) * 3072 + h * 64;
    #pragma unroll
    for (int kk = 0; kk < 2; ++kk) {
      qa[kk].q[0] = *(const uint4*)(qp + kk*32 + hl*8);
      qa[kk].q[1] = *(const uint4*)(qp + kk*32 + 16 + hl*8);
    }
  }

  const v8f vz = {0.f,0.f,0.f,0.f,0.f,0.f,0.f,0.f};
  v8f o[4];
  #pragma unroll
  for (int nj = 0; nj < 4; ++nj) o[nj] = vz;
  float mstat[8], lstat[8];
  #pragma unroll
  for (int r = 0; r < 8; ++r) { mstat[r] = -1.0e30f; lstat[r] = 0.f; }

  const int ktiles = q_base / 64 + 2;   // causal: k <= q_base+127
  for (int t = 0; t < ktiles; ++t) {
    const int k0g = t * 64;
    { // K tile async row-major; V tile transposed via VGPR
      int r   = tid >> 2;
      int seg = (tid & 3) * 16;
      const __bf16* kg = qkv + rowbase + (size_t)(k0g + r) * 3072 + 1024 + h*64 + seg;
      async_copy32(kg, Ks + r * 72 + seg);
      const __bf16* vg = qkv + rowbase + (size_t)(k0g + r) * 3072 + 2048 + h*64 + seg;
      uint4 v0 = ((const uint4*)vg)[0];
      uint4 v1 = ((const uint4*)vg)[1];
      const __bf16* e0 = (const __bf16*)&v0;
      const __bf16* e1 = (const __bf16*)&v1;
      #pragma unroll
      for (int i = 0; i < 8; ++i) Vs[(seg + i)     * 72 + r] = e0[i];
      #pragma unroll
      for (int i = 0; i < 8; ++i) Vs[(seg + 8 + i) * 72 + r] = e1[i];
    }
    wait_async0();
    __syncthreads();

    // S = Q @ K^T  (B frag = contiguous K rows)
    v8f s[4];
    #pragma unroll
    for (int nj = 0; nj < 4; ++nj) s[nj] = vz;
    #pragma unroll
    for (int nj = 0; nj < 4; ++nj) {
      const __bf16* p = Ks + (nj*16 + l16) * 72;
      #pragma unroll
      for (int kk = 0; kk < 2; ++kk) {
        FragB16 bf;
        bf.q[0] = *(const uint4*)(p + kk*32 + hl*8);
        bf.q[1] = *(const uint4*)(p + kk*32 + 16 + hl*8);
        s[nj] = wmma_bf16(qa[kk].v, bf.v, s[nj]);
      }
    }

    float am[4]; int kp[4];
    #pragma unroll
    for (int nj = 0; nj < 4; ++nj) {
      kp[nj] = k0g + nj*16 + l16;
      am[nj] = (1.f - amask[b * 2048 + kp[nj]]) * -10000.f;
    }

    // online softmax per q row (row lives in a 16-lane half)
    #pragma unroll
    for (int r = 0; r < 8; ++r) {
      const int qrow = q0 + r + hl * 8;
      float pv[4], rowmax = -3.0e30f;
      #pragma unroll
      for (int nj = 0; nj < 4; ++nj) {
        float sv = s[nj][r] * 0.125f;          // 1/sqrt(64)
        if (kp[nj] > qrow) sv = -10000.f;      // causal replace
        sv += am[nj];                          // additive padding mask
        pv[nj] = sv;
        rowmax = fmaxf(rowmax, sv);
      }
      #pragma unroll
      for (int off = 8; off >= 1; off >>= 1)
        rowmax = fmaxf(rowmax, __shfl_xor(rowmax, off, 32));
      float mn = fmaxf(mstat[r], rowmax);
      float corr = __expf(mstat[r] - mn);
      float rsum = 0.f;
      #pragma unroll
      for (int nj = 0; nj < 4; ++nj) {
        float e = __expf(pv[nj] - mn);
        pv[nj] = e; rsum += e;
      }
      #pragma unroll
      for (int off = 8; off >= 1; off >>= 1)
        rsum += __shfl_xor(rsum, off, 32);
      lstat[r] = lstat[r] * corr + rsum;
      mstat[r] = mn;
      #pragma unroll
      for (int nj = 0; nj < 4; ++nj) {
        o[nj][r] *= corr;
        Ss[wv][(r + hl*8) * 72 + nj*16 + l16] = (__bf16)pv[nj];
      }
    }

    // O += P @ V  (P A-frags from per-wave LDS strip; V B-frags from Vs)
    #pragma unroll
    for (int kk = 0; kk < 2; ++kk) {
      FragB16 pa;
      const __bf16* pp = &Ss[wv][l16 * 72 + kk * 32];
      pa.q[0] = *(const uint4*)(pp + hl*8);
      pa.q[1] = *(const uint4*)(pp + 16 + hl*8);
      #pragma unroll
      for (int nj = 0; nj < 4; ++nj) {
        const __bf16* p = Vs + (nj*16 + l16) * 72 + kk*32;
        FragB16 vb;
        vb.q[0] = *(const uint4*)(p + hl*8);
        vb.q[1] = *(const uint4*)(p + 16 + hl*8);
        o[nj] = wmma_bf16(pa.v, vb.v, o[nj]);
      }
    }
    __syncthreads();
  }

  #pragma unroll
  for (int r = 0; r < 8; ++r) {
    float inv = 1.f / lstat[r];
    int grow = b * 2048 + q0 + r + hl * 8;
    #pragma unroll
    for (int nj = 0; nj < 4; ++nj)
      ybuf[(size_t)grow * 1024 + h*64 + nj*16 + l16] = (__bf16)(o[nj][r] * inv);
  }
}

// ---------------------------------------------------------------------------
// LayerNorm over C=1024, one 128-thread block per row.
// ---------------------------------------------------------------------------
template<bool OUT_BF16>
__global__ __launch_bounds__(128)
void ln_kernel(const float* __restrict__ x, const float* __restrict__ g,
               const float* __restrict__ bb, void* __restrict__ outp)
{
  __shared__ float red[4], red2[4];
  const int row = blockIdx.x, tid = threadIdx.x;
  const int lane = tid & 31, wv = tid >> 5;
  const float* xr = x + (size_t)row * 1024;
  float4 a0 = ((const float4*)xr)[tid * 2];
  float4 a1 = ((const float4*)xr)[tid * 2 + 1];
  float v[8] = {a0.x,a0.y,a0.z,a0.w, a1.x,a1.y,a1.z,a1.w};
  float s = 0.f;
  #pragma unroll
  for (int i = 0; i < 8; ++i) s += v[i];
  #pragma unroll
  for (int off = 16; off >= 1; off >>= 1) s += __shfl_xor(s, off, 32);
  if (lane == 0) red[wv] = s;
  __syncthreads();
  float mu = (red[0] + red[1] + red[2] + red[3]) * (1.f / 1024.f);
  float vs = 0.f;
  #pragma unroll
  for (int i = 0; i < 8; ++i) { float d = v[i] - mu; vs += d * d; }
  #pragma unroll
  for (int off = 16; off >= 1; off >>= 1) vs += __shfl_xor(vs, off, 32);
  if (lane == 0) red2[wv] = vs;
  __syncthreads();
  float var = (red2[0] + red2[1] + red2[2] + red2[3]) * (1.f / 1024.f);
  float rs = rsqrtf(var + 1e-5f);
  #pragma unroll
  for (int i = 0; i < 8; ++i) {
    int col = tid * 8 + i;
    float o = (v[i] - mu) * rs * g[col] + bb[col];
    if (OUT_BF16) ((__bf16*)outp)[(size_t)row * 1024 + col] = (__bf16)o;
    else          ((float*)outp)[(size_t)row * 1024 + col] = o;
  }
}

// ---------------------------------------------------------------------------
extern "C" void kernel_launch(void* const* d_in, const int* in_sizes, int n_in,
                              void* d_out, int out_size, void* d_ws, size_t ws_size,
                              hipStream_t stream)
{
  (void)in_sizes; (void)n_in; (void)out_size; (void)ws_size;
  const int Lc = 8, Cc = 1024, MT = 4096;  // MT = B*T

  const float* embeds = (const float*)d_in[0];
  const float* amask  = (const float*)d_in[1];
  const float* ln1_g  = (const float*)d_in[2];
  const float* ln1_b  = (const float*)d_in[3];
  const float* attn_w = (const float*)d_in[4];
  const float* attn_b = (const float*)d_in[5];
  const float* proj_w = (const float*)d_in[6];
  const float* proj_b = (const float*)d_in[7];
  const float* ln2_g  = (const float*)d_in[8];
  const float* ln2_b  = (const float*)d_in[9];
  const float* fc_w   = (const float*)d_in[10];
  const float* fc_b   = (const float*)d_in[11];
  const float* p2_w   = (const float*)d_in[12];
  const float* p2_b   = (const float*)d_in[13];
  const float* lnf_g  = (const float*)d_in[14];
  const float* lnf_b  = (const float*)d_in[15];

  // workspace layout (~113 MB total)
  char* w = (char*)d_ws;
  float*  xbuf = (float*)w;   w += (size_t)MT * Cc * 4;        // residual f32
  __bf16* hbuf = (__bf16*)w;  w += (size_t)MT * Cc * 2;        // LN out bf16
  __bf16* qkvb = (__bf16*)w;  w += (size_t)MT * 3 * Cc * 2;    // qkv bf16
  __bf16* ybuf = (__bf16*)w;  w += (size_t)MT * Cc * 2;        // attn out bf16
  __bf16* mbuf = (__bf16*)w;  w += (size_t)MT * 4 * Cc * 2;    // gelu(fc) bf16
  __bf16* wtA  = (__bf16*)w;  w += (size_t)3 * Cc * Cc * 2;    // attn_w^T bf16
  __bf16* wtP  = (__bf16*)w;  w += (size_t)Cc * Cc * 2;        // proj_w^T
  __bf16* wtF  = (__bf16*)w;  w += (size_t)4 * Cc * Cc * 2;    // fc_w^T
  __bf16* wtP2 = (__bf16*)w;  w += (size_t)4 * Cc * Cc * 2;    // proj2_w^T

  const dim3 blk(256);
  const size_t gemm_lds = (size_t)4 * TILE_HALVES * sizeof(__bf16);  // 73728 B
  const float* xsrc = embeds;
  for (int l = 0; l < Lc; ++l) {
    // pre-transpose/convert this layer's weights to bf16 [N,K]
    wtrans_kernel<<<dim3(48, 16), blk, 0, stream>>>(
        attn_w + (size_t)l*Cc*3*Cc, wtA, Cc, 3*Cc);
    wtrans_kernel<<<dim3(16, 16), blk, 0, stream>>>(
        proj_w + (size_t)l*Cc*Cc, wtP, Cc, Cc);
    wtrans_kernel<<<dim3(64, 16), blk, 0, stream>>>(
        fc_w + (size_t)l*Cc*4*Cc, wtF, Cc, 4*Cc);
    wtrans_kernel<<<dim3(16, 64), blk, 0, stream>>>(
        p2_w + (size_t)l*4*Cc*Cc, wtP2, 4*Cc, Cc);

    ln_kernel<true><<<MT, 128, 0, stream>>>(xsrc, ln1_g + l*Cc, ln1_b + l*Cc, hbuf);

    gemm_kernel<false,false,false,true><<<dim3(24, 32), blk, gemm_lds, stream>>>(
        hbuf, wtA, attn_b + (size_t)l*3*Cc, nullptr, nullptr, qkvb, MT, 3*Cc, Cc);

    attn_kernel<<<dim3(16, 32), blk, 0, stream>>>(qkvb, amask, ybuf);

    gemm_kernel<false,true,true,false><<<dim3(8, 32), blk, gemm_lds, stream>>>(
        ybuf, wtP, proj_b + (size_t)l*Cc, xsrc, xbuf, nullptr, MT, Cc, Cc);

    ln_kernel<true><<<MT, 128, 0, stream>>>(xbuf, ln2_g + l*Cc, ln2_b + l*Cc, hbuf);

    gemm_kernel<true,false,false,true><<<dim3(32, 32), blk, gemm_lds, stream>>>(
        hbuf, wtF, fc_b + (size_t)l*4*Cc, nullptr, nullptr, mbuf, MT, 4*Cc, Cc);

    gemm_kernel<false,true,true,false><<<dim3(8, 32), blk, gemm_lds, stream>>>(
        mbuf, wtP2, p2_b + (size_t)l*Cc, xbuf, xbuf, nullptr, MT, Cc, 4*Cc);

    xsrc = xbuf;
  }
  ln_kernel<false><<<MT, 128, 0, stream>>>(xbuf, lnf_g, lnf_b, d_out);
}
